// CrossModalAttention_20229295964778
// MI455X (gfx1250) — matrix operand order repaired and checked by reading, not compiled, not run on hardware
//
#include <hip/hip_runtime.h>

// ---------------------------------------------------------------------------
// CrossModalAttention for MI455X (gfx1250, wave32, WMMA f32_16x16x32_f16)
//   B=8, Cc=128, Ce=256, D=128, N=64*64=4096
//   ~75 GFLOP vs ~70 MB traffic -> deeply compute bound: f16 WMMA everywhere,
//   f32 accumulate, async-to-LDS double-buffered K/V staging in attention.
// ---------------------------------------------------------------------------

typedef __attribute__((ext_vector_type(16))) _Float16 v16h;
typedef __attribute__((ext_vector_type(8)))  _Float16 v8h;
typedef __attribute__((ext_vector_type(8)))  float    v8f;
typedef int v4i_vs __attribute__((vector_size(16)));   // matches builtin proto

#define CAPE_DIM 128
#define ERA5_DIM 256
#define ATTN_DIM 128
#define BATCH    8
#define NPIX     4096
#define QK_SCALE 0.08838834764831845f   // 128^-0.5, folded into Q

// gfx1250 async global->LDS path (ASYNCcnt-tracked). Guarded so the file
// still compiles (with a synchronous fallback) if the toolchain lacks them.
#if defined(__HIP_DEVICE_COMPILE__) &&                                  \
    __has_builtin(__builtin_amdgcn_global_load_async_to_lds_b128) &&    \
    __has_builtin(__builtin_amdgcn_s_wait_asynccnt)
#define USE_ASYNC_LDS 1
#else
#define USE_ASYNC_LDS 0
#endif

__device__ __forceinline__ void copy16B_g2l(_Float16* lds_dst,
                                            const _Float16* gsrc) {
#if USE_ASYNC_LDS
  // (global v4i* src, lds v4i* dst, imm offset, imm cpol); clang rewrites the
  // builtin's pointer address spaces to match the (generic) arguments.
  __builtin_amdgcn_global_load_async_to_lds_b128(
      (v4i_vs*)gsrc, (v4i_vs*)lds_dst, 0, 0);
#else
  *(v8h*)lds_dst = *(const v8h*)gsrc;
#endif
}

__device__ __forceinline__ void wait_async_le8() {
#if USE_ASYNC_LDS
  __builtin_amdgcn_s_wait_asynccnt(8);
#endif
}
__device__ __forceinline__ void wait_async_0() {
#if USE_ASYNC_LDS
  __builtin_amdgcn_s_wait_asynccnt(0);
#endif
}

__device__ __forceinline__ v8f wmma_f16(v16h a, v16h b, v8f c) {
  // 8 args: (neg_a, A, neg_b, B, c_mod, C, reuse_a, reuse_b)
  return __builtin_amdgcn_wmma_f32_16x16x32_f16(false, a, false, b,
                                                (short)0, c, false, false);
}

// A fragment (16x32, M=rows, K=cols) from f32 row-major matrix, converted.
// Lane (half=lane>>4, m=lane&15) holds K = kbase + half*16 + {0..15}.
__device__ __forceinline__ v16h loadA_f32(const float* p, int ld,
                                          int mbase, int kbase, int lane) {
  const int row  = mbase + (lane & 15);
  const int half = lane >> 4;
  const float* r0 = p + (size_t)row * ld + kbase + half * 16;
  v16h a;
#pragma unroll
  for (int i = 0; i < 8; ++i) {
    a[i]     = (_Float16)r0[i];
    a[8 + i] = (_Float16)r0[8 + i];
  }
  return a;
}

// A fragment from f16 storage (global or LDS) where M-rows are contiguous in K.
__device__ __forceinline__ v16h loadA_f16(const _Float16* p, int ld,
                                          int mbase, int kbase, int lane) {
  const int row  = mbase + (lane & 15);
  const int half = lane >> 4;
  const _Float16* r0 = p + (size_t)row * ld + kbase + half * 16;
  v8h lo = *(const v8h*)r0;          // 16B
  v8h hi = *(const v8h*)(r0 + 8);    // 16B
  v16h a;
#pragma unroll
  for (int i = 0; i < 8; ++i) { a[i] = lo[i]; a[8 + i] = hi[i]; }
  return a;
}

// B fragment (32x16, K=rows, N=cols) from f32 matrix stored [K][N] (stride ld).
__device__ __forceinline__ v16h loadB_f32_strided(const float* p, int ld,
                                                  int kbase, int nbase,
                                                  int lane) {
  const int col = nbase + (lane & 15);
  const int k0  = kbase + (lane >> 4) * 16;
  v16h b;
#pragma unroll
  for (int i = 0; i < 16; ++i)
    b[i] = (_Float16)p[(size_t)(k0 + i) * ld + col];
  return b;
}

// B fragment from f16 storage laid out [N][K] (K contiguous, stride ld):
// one 32-byte contiguous load per lane.
__device__ __forceinline__ v16h loadB_f16(const _Float16* p, int ld,
                                          int kbase, int nbase, int lane) {
  const _Float16* r0 =
      p + (size_t)(nbase + (lane & 15)) * ld + kbase + (lane >> 4) * 16;
  return *(const v16h*)r0;           // 32B
}

// ---------------------------------------------------------------------------
// Kernel 1: fused QKV projection (1x1 convs == GEMMs), f32 -> f16 tile store.
//   blockIdx.y selects matrix (0=Q,1=K,2=V), blockIdx.z = batch.
//   Q,K stored [b][n][d]; V stored [b][d][n].
// ---------------------------------------------------------------------------
__global__ __launch_bounds__(128) void k_qkv(
    const float* __restrict__ cape, const float* __restrict__ era5,
    const float* __restrict__ Wq, const float* __restrict__ bq,
    const float* __restrict__ Wk, const float* __restrict__ bk,
    const float* __restrict__ Wv, const float* __restrict__ bv,
    _Float16* __restrict__ Qh, _Float16* __restrict__ Kh,
    _Float16* __restrict__ Vh) {
  const int lane  = threadIdx.x & 31;
  const int wave  = threadIdx.x >> 5;
  const int b     = blockIdx.z;
  const int which = blockIdx.y;
  const int n0    = blockIdx.x * 64 + wave * 16;

  const float* W;
  const float* bias;
  const float* src;
  int Kd;
  if (which == 0) {
    W = Wq; bias = bq; src = cape + (size_t)b * CAPE_DIM * NPIX; Kd = CAPE_DIM;
  } else if (which == 1) {
    W = Wk; bias = bk; src = era5 + (size_t)b * ERA5_DIM * NPIX; Kd = ERA5_DIM;
  } else {
    W = Wv; bias = bv; src = era5 + (size_t)b * ERA5_DIM * NPIX; Kd = ERA5_DIM;
  }

  v8f acc[8] = {};
  for (int k0 = 0; k0 < Kd; k0 += 32) {
    v16h bf = loadB_f32_strided(src, NPIX, k0, n0, lane);  // activations
#pragma unroll
    for (int mt = 0; mt < 8; ++mt) {
      v16h af = loadA_f32(W, Kd, mt * 16, k0, lane);        // weights
      acc[mt] = wmma_f16(af, bf, acc[mt]);
    }
  }

  const int half = lane >> 4;
  const int col  = lane & 15;
  const float sc = (which == 0) ? QK_SCALE : 1.0f;
#pragma unroll
  for (int mt = 0; mt < 8; ++mt) {
#pragma unroll
    for (int r = 0; r < 8; ++r) {
      const int d = mt * 16 + r + 8 * half;   // C layout: M = r + 8*half
      const int n = n0 + col;                 //           N = lane&15
      const _Float16 hv = (_Float16)((acc[mt][r] + bias[d]) * sc);
      if (which == 0)
        Qh[((size_t)b * NPIX + n) * ATTN_DIM + d] = hv;
      else if (which == 1)
        Kh[((size_t)b * NPIX + n) * ATTN_DIM + d] = hv;
      else
        Vh[((size_t)b * ATTN_DIM + d) * NPIX + n] = hv;
    }
  }
}

// ---------------------------------------------------------------------------
// Kernel 2: flash attention with double-buffered async K/V staging in LDS.
//   Block = 4 waves; each wave owns one 16-query tile; the block shares
//   the K/V stream (32 keys per step, staged once via async-to-LDS).
// ---------------------------------------------------------------------------
#define KSTEPS (NPIX / 32)

__global__ __launch_bounds__(128) void k_attn(
    const _Float16* __restrict__ Qh, const _Float16* __restrict__ Kh,
    const _Float16* __restrict__ Vh, _Float16* __restrict__ Oh) {
  __shared__ __align__(32) _Float16 Ks[2][32 * ATTN_DIM];  // [kn][d]  8KB x2
  __shared__ __align__(32) _Float16 Vs[2][ATTN_DIM * 32];  // [d][kn]  8KB x2
  __shared__ __align__(32) _Float16 ldsP[4][16 * 32];      // per-wave P [qn][kn]
  __shared__ float ldsStat[4][16];                         // per-wave row stats

  const int tid  = threadIdx.x;
  const int lane = tid & 31;
  const int wave = tid >> 5;
  const int b    = blockIdx.y;
  const int q0   = (blockIdx.x * 4 + wave) * 16;
  const int half = lane >> 4;
  const int col  = lane & 15;

  const _Float16* Qb = Qh + (size_t)b * NPIX * ATTN_DIM;
  const _Float16* Kb = Kh + (size_t)b * NPIX * ATTN_DIM;
  const _Float16* Vb = Vh + (size_t)b * ATTN_DIM * NPIX;
  _Float16* Pw = ldsP[wave];
  float*    Sw = ldsStat[wave];

  // Resident Q fragments: A layout, M=qn, K=d (4 x 32).
  v16h qf[4];
#pragma unroll
  for (int ks = 0; ks < 4; ++ks)
    qf[ks] = loadA_f16(Qb, ATTN_DIM, q0, ks * 32, lane);

  v8f o[8] = {};
  float mrow[8], lrow[8];
#pragma unroll
  for (int r = 0; r < 8; ++r) { mrow[r] = -1e30f; lrow[r] = 0.0f; }

  // --- stage kt=0 slabs: each of 128 threads copies 8 x 16B --------------
  // K slab: 32 rows x 128 d, contiguous 8KB.  V slab: 128 rows x 32 kn.
  {
    const _Float16* gk = Kb + (size_t)0 * ATTN_DIM + tid * 32;
    const _Float16* gv = Vb + (size_t)tid * NPIX + 0;
#pragma unroll
    for (int j = 0; j < 4; ++j) copy16B_g2l(&Ks[0][tid * 32 + j * 8], gk + j * 8);
#pragma unroll
    for (int j = 0; j < 4; ++j) copy16B_g2l(&Vs[0][tid * 32 + j * 8], gv + j * 8);
  }

  for (int kt = 0; kt < KSTEPS; ++kt) {
    const int cur = kt & 1;
    __syncthreads();  // all waves done reading buf[1-cur] (prev compute)

    if (kt + 1 < KSTEPS) {
      const int kn1 = (kt + 1) * 32;
      const _Float16* gk = Kb + (size_t)kn1 * ATTN_DIM + tid * 32;
      const _Float16* gv = Vb + (size_t)tid * NPIX + kn1;
#pragma unroll
      for (int j = 0; j < 4; ++j)
        copy16B_g2l(&Ks[1 - cur][tid * 32 + j * 8], gk + j * 8);
#pragma unroll
      for (int j = 0; j < 4; ++j)
        copy16B_g2l(&Vs[1 - cur][tid * 32 + j * 8], gv + j * 8);
      wait_async_le8();  // our 8 ops for buf[cur] retired; new 8 in flight
    } else {
      wait_async_0();
    }
    __syncthreads();  // every wave's slab portion landed in buf[cur]

    const _Float16* Kc = Ks[cur];   // [kn 0..31][d 0..127]
    const _Float16* Vc = Vs[cur];   // [d 0..127][kn 0..31]

    // ---- S = Q^T K : two 16x16 tiles covering 32 keys -------------------
    v8f s0 = {}, s1 = {};
#pragma unroll
    for (int ks = 0; ks < 4; ++ks) {
      v16h kb0 = loadB_f16(Kc, ATTN_DIM, ks * 32, 0, lane);
      v16h kb1 = loadB_f16(Kc, ATTN_DIM, ks * 32, 16, lane);
      s0 = wmma_f16(qf[ks], kb0, s0);
      s1 = wmma_f16(qf[ks], kb1, s1);
    }

    // ---- online softmax (rows live across 16-lane half-groups) ----------
    float fac[8];
#pragma unroll
    for (int r = 0; r < 8; ++r) {
      float mx = fmaxf(s0[r], s1[r]);
#pragma unroll
      for (int off = 1; off < 16; off <<= 1)
        mx = fmaxf(mx, __shfl_xor(mx, off));
      const float nm = fmaxf(mrow[r], mx);
      const float p0 = __expf(s0[r] - nm);
      const float p1 = __expf(s1[r] - nm);
      float rs = p0 + p1;
#pragma unroll
      for (int off = 1; off < 16; off <<= 1)
        rs += __shfl_xor(rs, off);
      const float f = __expf(mrow[r] - nm);
      lrow[r] = lrow[r] * f + rs;
      mrow[r] = nm;
      fac[r]  = f;
      // Stage P transposed for the V*P^T B fragment: LDS[qn][kn].
      const int qn = r + 8 * half;
      Pw[qn * 32 + col]      = (_Float16)p0;
      Pw[qn * 32 + 16 + col] = (_Float16)p1;
    }

    // Bridge per-row rescale factor to per-qn lanes via LDS (wave-private
    // region, LDS pipe is in-order within a wave -> no barrier needed).
    if (col == 0) {
#pragma unroll
      for (int r = 0; r < 8; ++r) Sw[r + 8 * half] = fac[r];
    }
    const float fq = Sw[col];
#pragma unroll
    for (int dt = 0; dt < 8; ++dt)
#pragma unroll
      for (int r = 0; r < 8; ++r) o[dt][r] *= fq;

    // ---- O += V * P^T ---------------------------------------------------
    const v16h pb = *(const v16h*)&Pw[col * 32 + half * 16];  // B: K=kn, N=qn
#pragma unroll
    for (int dt = 0; dt < 8; ++dt) {
      v16h vf = loadA_f16(Vc, 32, dt * 16, 0, lane);          // A: M=d, K=kn
      o[dt] = wmma_f16(vf, pb, o[dt]);
    }
  }

  // Final 1/l normalization (per-qn, bridged through LDS) and store [b][n][d].
  if (col == 0) {
#pragma unroll
    for (int r = 0; r < 8; ++r) Sw[r + 8 * half] = lrow[r];
  }
  const float linv = 1.0f / Sw[col];
#pragma unroll
  for (int dt = 0; dt < 8; ++dt) {
#pragma unroll
    for (int r = 0; r < 8; ++r) {
      const int d = dt * 16 + r + 8 * half;
      Oh[((size_t)b * NPIX + q0 + col) * ATTN_DIM + d] =
          (_Float16)(o[dt][r] * linv);
    }
  }
}

// ---------------------------------------------------------------------------
// Kernel 3: output projection out = Wo * O + bo, f32 result in [b][c][n].
// ---------------------------------------------------------------------------
__global__ __launch_bounds__(128) void k_oproj(
    const _Float16* __restrict__ Oh, const float* __restrict__ Wo,
    const float* __restrict__ bo, float* __restrict__ out) {
  const int lane = threadIdx.x & 31;
  const int wave = threadIdx.x >> 5;
  const int b    = blockIdx.y;
  const int n0   = blockIdx.x * 64 + wave * 16;
  const _Float16* Ob = Oh + (size_t)b * NPIX * ATTN_DIM;

  v8f acc[8] = {};
#pragma unroll
  for (int k0 = 0; k0 < ATTN_DIM; k0 += 32) {
    v16h bf = loadB_f16(Ob, ATTN_DIM, k0, n0, lane);
#pragma unroll
    for (int mt = 0; mt < 8; ++mt) {
      v16h af = loadA_f32(Wo, ATTN_DIM, mt * 16, k0, lane);
      acc[mt] = wmma_f16(af, bf, acc[mt]);
    }
  }

  const int half = lane >> 4;
  const int col  = lane & 15;
#pragma unroll
  for (int mt = 0; mt < 8; ++mt) {
#pragma unroll
    for (int r = 0; r < 8; ++r) {
      const int oc = mt * 16 + r + 8 * half;
      out[(size_t)b * CAPE_DIM * NPIX + (size_t)oc * NPIX + n0 + col] =
          acc[mt][r] + bo[oc];
    }
  }
}

// ---------------------------------------------------------------------------
extern "C" void kernel_launch(void* const* d_in, const int* in_sizes, int n_in,
                              void* d_out, int out_size, void* d_ws,
                              size_t ws_size, hipStream_t stream) {
  const float* cape = (const float*)d_in[0];
  const float* era5 = (const float*)d_in[1];
  const float* Wq   = (const float*)d_in[2];
  const float* bq   = (const float*)d_in[3];
  const float* Wk   = (const float*)d_in[4];
  const float* bk   = (const float*)d_in[5];
  const float* Wv   = (const float*)d_in[6];
  const float* bv   = (const float*)d_in[7];
  const float* Wo   = (const float*)d_in[8];
  const float* bo   = (const float*)d_in[9];
  float* out = (float*)d_out;

  const size_t SZ = (size_t)BATCH * NPIX * ATTN_DIM;  // 4,194,304 f16 = 8 MB
  _Float16* ws = (_Float16*)d_ws;
  _Float16* Qh = ws;            // [b][n][d]  (pre-scaled by 128^-0.5)
  _Float16* Kh = ws + SZ;       // [b][n][d]
  _Float16* Vh = ws + 2 * SZ;   // [b][d][n]
  _Float16* Oh = ws + 3 * SZ;   // [b][n][d]

  // 1) QKV projection: grid (N/64 slabs, {Q,K,V}, batch)
  k_qkv<<<dim3(NPIX / 64, 3, BATCH), 128, 0, stream>>>(
      cape, era5, Wq, bq, Wk, bk, Wv, bv, Qh, Kh, Vh);

  // 2) Flash attention: 256 query tiles / batch, 4 tiles (waves) per block
  k_attn<<<dim3(NPIX / 16 / 4, BATCH), 128, 0, stream>>>(Qh, Kh, Vh, Oh);

  // 3) Output projection
  k_oproj<<<dim3(NPIX / 64, BATCH), 128, 0, stream>>>(Oh, Wo, bo, out);
}
